// EnhancedMultiheadAttention_18210661335360
// MI455X (gfx1250) — matrix operand order repaired and checked by reading, not compiled, not run on hardware
//
#include <hip/hip_runtime.h>
#include <hip/hip_bf16.h>
#include <math.h>

// ---------------------------------------------------------------------------
// EnhancedMultiheadAttention forward for MI455X (gfx1250, wave32, WMMA).
// B=2, S=2048, D_MODEL=1024, H=16, hd=64.
// d_out = out [B,S,D] fp32  ++  attn_weights [B,H,S,S] fp32.
// ---------------------------------------------------------------------------

#define D_MODEL   1024
#define NUM_HEADS 16
#define HEAD_DIM  64
#define BATCH     2
#define SEQ       2048
#define ROWS      (BATCH * SEQ)          // 4096 GEMM rows for projections

typedef __bf16  bf16_t;
typedef __attribute__((ext_vector_type(16))) __bf16 v16bf;
typedef __attribute__((ext_vector_type(8)))  float  v8f;

// fp32 -> bf16 round-to-nearest-even (elementwise kernels / epilogues only)
static __device__ __forceinline__ bf16_t f2bf(float f) {
    union { float f; unsigned u; } in; in.f = f;
    unsigned r = in.u + 0x7FFFu + ((in.u >> 16) & 1u);
    union { unsigned short s; bf16_t b; } out; out.s = (unsigned short)(r >> 16);
    return out.b;
}

// ---- WMMA fragment index maps (16x16x32 bf16, wave32, per CDNA5 ISA) -------
// A 16x32 (MxK): lane<16 -> row=lane,    K = {k0+0..7,  k0+16..23}
//                lane>=16 -> row=lane-16, K = {k0+8..15, k0+24..31}
static __device__ __forceinline__ int a_k(int j, int half) {
    return half * 8 + ((j < 8) ? j : (j + 8));
}
// B 32x16 (KxN): lane holds column N=lane&15; same K order as A:
// lanes 0-15 -> K={k0+0..7, k0+16..23} ; lanes 16-31 -> K={k0+8..15, k0+24..31}.
static __device__ __forceinline__ int b_k(int j, int half) {
    return half * 8 + ((j < 8) ? j : (j + 8));
}

static __device__ __forceinline__ v16bf load_a_bf16(const bf16_t* base, int ld,
                                                    int row, int k0, int half) {
    v16bf a;
#pragma unroll
    for (int j = 0; j < 16; ++j)
        a[j] = base[(size_t)row * ld + k0 + a_k(j, half)];
    return a;
}

static __device__ __forceinline__ v16bf load_a_f32(const float* base, int ld,
                                                   int row, int k0, int half) {
    v16bf a;
#pragma unroll
    for (int j = 0; j < 16; ++j)
        a[j] = f2bf(base[(size_t)row * ld + k0 + a_k(j, half)]);
    return a;
}

// B stored [N][K] row-major (weights W[n][k], K-head rows, V^T): contiguous in k.
static __device__ __forceinline__ v16bf load_b_nk(const bf16_t* base, int ld,
                                                  int n, int k0, int half) {
    v16bf b;
#pragma unroll
    for (int j = 0; j < 16; ++j)
        b[j] = base[(size_t)n * ld + k0 + b_k(j, half)];
    return b;
}

static __device__ __forceinline__ v8f wmma_bf16(v16bf a, v16bf b, v8f c) {
    return __builtin_amdgcn_wmma_f32_16x16x32_bf16(
        false, a, false, b, (short)0, c, false, false);
}

#define ZERO8 v8f{0.f,0.f,0.f,0.f,0.f,0.f,0.f,0.f}

// ---------------------------------------------------------------------------
// Kernel 1: fp32 -> bf16 bulk convert (weights and activations)
// ---------------------------------------------------------------------------
__global__ void cvt_f32_bf16_kernel(const float* __restrict__ src,
                                    bf16_t* __restrict__ dst, int n) {
    int i = blockIdx.x * blockDim.x + threadIdx.x;
    if (i < n) dst[i] = f2bf(src[i]);
}

// ---------------------------------------------------------------------------
// Kernel 1b: per-(b,h) score scale factors (hoists sinf out of scores_kernel)
// sf[b*H+h] = 0.125 * attn_scale * (1 + coupling[h]*sin(phase[b]))
// ---------------------------------------------------------------------------
__global__ void score_scale_kernel(const float* __restrict__ phase,
                                   const float* __restrict__ coupling,
                                   const float* __restrict__ attn_scale,
                                   float* __restrict__ sf) {
    const int i = threadIdx.x;              // one wave, 32 = B*H entries
    const int b = i >> 4, h = i & 15;
    sf[i] = 0.125f * attn_scale[0] * (1.0f + coupling[h] * sinf(phase[b]));
}

// ---------------------------------------------------------------------------
// Kernel 2: projection  Y = X @ W^T + b.
// One wave computes a 32x64 slab (2 M-tiles x 4 N-tiles): 8 WMMAs per K-step
// against 6 fragment loads.
// MODE 0: head-split [B,H,S,hd] bf16 (Q,K).  MODE 1: transposed [B,H,hd,S] (V).
// ---------------------------------------------------------------------------
template <int MODE>
__global__ void proj_kernel(const bf16_t* __restrict__ X,    // [4096,1024] bf16
                            const bf16_t* __restrict__ Wbf,  // [1024,1024] bf16 W[n][k]
                            const float* __restrict__ bias,  // [1024]
                            bf16_t* __restrict__ dst) {
    const int wave = threadIdx.x >> 5;
    const int lane = threadIdx.x & 31;
    const int half = lane >> 4;
    const int lrow = lane & 15;

    const int group = blockIdx.x * 8 + wave;   // 128 M-groups * 16 N-groups
    const int tng = group & 15;                // N group (4 tiles)
    const int tmg = group >> 4;                // M group (2 tiles)

    const int aRow0 = tmg * 32 + lrow;
    const int aRow1 = aRow0 + 16;

    v8f acc[2][4];
#pragma unroll
    for (int m = 0; m < 2; ++m)
#pragma unroll
        for (int n = 0; n < 4; ++n) acc[m][n] = ZERO8;

    for (int k0 = 0; k0 < D_MODEL; k0 += 32) {
        __builtin_prefetch(&X[(size_t)aRow0 * D_MODEL + k0 + 256], 0, 3);
        v16bf a0 = load_a_bf16(X, D_MODEL, aRow0, k0, half);
        v16bf a1 = load_a_bf16(X, D_MODEL, aRow1, k0, half);
#pragma unroll
        for (int n = 0; n < 4; ++n) {
            const int bCol = (tng * 4 + n) * 16 + lrow;
            v16bf b = load_b_nk(Wbf, D_MODEL, bCol, k0, half);
            acc[0][n] = wmma_bf16(a0, b, acc[0][n]);
            acc[1][n] = wmma_bf16(a1, b, acc[1][n]);
        }
    }

#pragma unroll
    for (int mt = 0; mt < 2; ++mt) {
#pragma unroll
        for (int n = 0; n < 4; ++n) {
            const int ncol = (tng * 4 + n) * 16 + lrow;    // C/D: lane -> N
            const int h = ncol >> 6, d = ncol & 63;
            const float bn = bias[ncol];
#pragma unroll
            for (int v = 0; v < 8; ++v) {
                const int m = tmg * 32 + mt * 16 + v + 8 * half; // C/D: VGPR -> M
                const int bb = m >> 11, s = m & 2047;
                const bf16_t val = f2bf(acc[mt][n][v] + bn);
                if (MODE == 0)
                    dst[(((size_t)bb * NUM_HEADS + h) * SEQ + s) * HEAD_DIM + d] = val;
                else
                    dst[(((size_t)bb * NUM_HEADS + h) * HEAD_DIM + d) * SEQ + s] = val;
            }
        }
    }
}

// ---------------------------------------------------------------------------
// Kernel 3: scores = (Qh Kh^T) * sf[bh].
// One wave computes 32 q-rows x 64 k-cols (2x4 tiles); K = hd = 64 (2 steps).
// ---------------------------------------------------------------------------
__global__ void scores_kernel(const bf16_t* __restrict__ qh,
                              const bf16_t* __restrict__ kh,
                              const float* __restrict__ sf,   // [B*H]
                              float* __restrict__ scores) {   // [B,H,S,S]
    const int wave = threadIdx.x >> 5;
    const int lane = threadIdx.x & 31;
    const int half = lane >> 4;
    const int lrow = lane & 15;

    const int group = blockIdx.x * 8 + wave;  // B*H * 64 q-groups * 32 k-groups
    const int bh   = group >> 11;
    const int rest = group & 2047;
    const int tqg = rest >> 5;                // 64 q-groups (32 rows each)
    const int tkg = rest & 31;                // 32 k-groups (64 cols each)

    const bf16_t* Q = qh + (size_t)bh * SEQ * HEAD_DIM;
    const bf16_t* K = kh + (size_t)bh * SEQ * HEAD_DIM;
    const int aRow0 = tqg * 32 + lrow;
    const int aRow1 = aRow0 + 16;

    v8f acc[2][4];
#pragma unroll
    for (int m = 0; m < 2; ++m)
#pragma unroll
        for (int n = 0; n < 4; ++n) acc[m][n] = ZERO8;

#pragma unroll
    for (int k0 = 0; k0 < HEAD_DIM; k0 += 32) {
        v16bf a0 = load_a_bf16(Q, HEAD_DIM, aRow0, k0, half);
        v16bf a1 = load_a_bf16(Q, HEAD_DIM, aRow1, k0, half);
#pragma unroll
        for (int n = 0; n < 4; ++n) {
            const int bCol = (tkg * 4 + n) * 16 + lrow;   // row of K-head
            v16bf bf = load_b_nk(K, HEAD_DIM, bCol, k0, half);
            acc[0][n] = wmma_bf16(a0, bf, acc[0][n]);
            acc[1][n] = wmma_bf16(a1, bf, acc[1][n]);
        }
    }

    const float s = sf[bh];
#pragma unroll
    for (int mt = 0; mt < 2; ++mt) {
#pragma unroll
        for (int n = 0; n < 4; ++n) {
            const int kcol = (tkg * 4 + n) * 16 + lrow;
#pragma unroll
            for (int v = 0; v < 8; ++v) {
                const int qrow = tqg * 32 + mt * 16 + v + 8 * half;
                scores[((size_t)bh * SEQ + qrow) * SEQ + kcol] = acc[mt][n][v] * s;
            }
        }
    }
}

// ---------------------------------------------------------------------------
// Kernel 4: row softmax (2048 cols), in place; optionally emits bf16 copy.
// ---------------------------------------------------------------------------
__global__ void softmax_kernel(float* __restrict__ attn,
                               bf16_t* __restrict__ attn_bf /* may be null */) {
    __shared__ float sred[256];
    const int tid = threadIdx.x;
    const size_t rowOff = (size_t)blockIdx.x * SEQ;
    float* row = attn + rowOff;

    float x[8];
#pragma unroll
    for (int i = 0; i < 8; ++i) x[i] = row[tid + i * 256];

    float mx = -3.0e38f;
#pragma unroll
    for (int i = 0; i < 8; ++i) mx = fmaxf(mx, x[i]);
    sred[tid] = mx; __syncthreads();
    for (int s = 128; s > 0; s >>= 1) {
        if (tid < s) sred[tid] = fmaxf(sred[tid], sred[tid + s]);
        __syncthreads();
    }
    mx = sred[0]; __syncthreads();

    float sum = 0.f;
#pragma unroll
    for (int i = 0; i < 8; ++i) { x[i] = __expf(x[i] - mx); sum += x[i]; }
    sred[tid] = sum; __syncthreads();
    for (int s = 128; s > 0; s >>= 1) {
        if (tid < s) sred[tid] += sred[tid + s];
        __syncthreads();
    }
    const float inv = 1.0f / sred[0];

    if (attn_bf != nullptr) {
#pragma unroll
        for (int i = 0; i < 8; ++i) {
            const float w = x[i] * inv;
            row[tid + i * 256] = w;
            attn_bf[rowOff + tid + i * 256] = f2bf(w);
        }
    } else {
#pragma unroll
        for (int i = 0; i < 8; ++i) row[tid + i * 256] = x[i] * inv;
    }
}

// ---------------------------------------------------------------------------
// Kernel 5: context = attn @ Vh.  One wave: 32 rows x full hd (2x4 tiles).
// B operand is V^T [B,H,hd,S] -> contiguous-in-k loads. Output ctx bf16 [B,S,D].
// ---------------------------------------------------------------------------
template <bool BF16A>
__global__ void context_kernel(const float* __restrict__ attn,     // [B,H,S,S] f32
                               const bf16_t* __restrict__ attn_bf, // same, bf16 (opt)
                               const bf16_t* __restrict__ vhT,     // [B,H,hd,S]
                               bf16_t* __restrict__ ctx) {         // [B,S,D] bf16
    const int wave = threadIdx.x >> 5;
    const int lane = threadIdx.x & 31;
    const int half = lane >> 4;
    const int lrow = lane & 15;

    const int group = blockIdx.x * 8 + wave;   // B*H * 64 M-groups
    const int bh  = group >> 6;
    const int tmg = group & 63;
    const int b = bh >> 4, h = bh & 15;

    const float*  Af = attn    + (size_t)bh * SEQ * SEQ;
    const bf16_t* Ab = attn_bf + (size_t)bh * SEQ * SEQ;
    const bf16_t* V  = vhT     + (size_t)bh * HEAD_DIM * SEQ;
    const int aRow0 = tmg * 32 + lrow;
    const int aRow1 = aRow0 + 16;

    v8f acc[2][4];
#pragma unroll
    for (int m = 0; m < 2; ++m)
#pragma unroll
        for (int n = 0; n < 4; ++n) acc[m][n] = ZERO8;

    for (int k0 = 0; k0 < SEQ; k0 += 32) {
        v16bf a0, a1;
        if (BF16A) {
            __builtin_prefetch(&Ab[(size_t)aRow0 * SEQ + k0 + 256], 0, 3);
            a0 = load_a_bf16(Ab, SEQ, aRow0, k0, half);
            a1 = load_a_bf16(Ab, SEQ, aRow1, k0, half);
        } else {
            __builtin_prefetch(&Af[(size_t)aRow0 * SEQ + k0 + 128], 0, 3);
            a0 = load_a_f32(Af, SEQ, aRow0, k0, half);
            a1 = load_a_f32(Af, SEQ, aRow1, k0, half);
        }
#pragma unroll
        for (int n = 0; n < 4; ++n) {
            const int bCol = n * 16 + lrow;     // d index
            v16bf bf = load_b_nk(V, SEQ, bCol, k0, half);
            acc[0][n] = wmma_bf16(a0, bf, acc[0][n]);
            acc[1][n] = wmma_bf16(a1, bf, acc[1][n]);
        }
    }

#pragma unroll
    for (int mt = 0; mt < 2; ++mt) {
#pragma unroll
        for (int n = 0; n < 4; ++n) {
            const int d = n * 16 + lrow;
#pragma unroll
            for (int v = 0; v < 8; ++v) {
                const int s = tmg * 32 + mt * 16 + v + 8 * half;
                ctx[((size_t)b * SEQ + s) * D_MODEL + h * HEAD_DIM + d] =
                    f2bf(acc[mt][n][v]);
            }
        }
    }
}

// ---------------------------------------------------------------------------
// Kernel 6: out = (ctx @ Wo^T + bo) * stability_factor, fp32 output.
// One wave computes 32x64; A is bf16 ctx.
// ---------------------------------------------------------------------------
__global__ void out_proj_kernel(const bf16_t* __restrict__ ctx,  // [4096,1024] bf16
                                const bf16_t* __restrict__ Wo,   // [1024,1024] bf16
                                const float* __restrict__ bo,    // [1024]
                                const float* __restrict__ stab,  // [1]
                                float* __restrict__ out) {       // [4096,1024]
    const int wave = threadIdx.x >> 5;
    const int lane = threadIdx.x & 31;
    const int half = lane >> 4;
    const int lrow = lane & 15;

    const int group = blockIdx.x * 8 + wave;   // 128 M-groups * 16 N-groups
    const int tng = group & 15;
    const int tmg = group >> 4;
    const int aRow0 = tmg * 32 + lrow;
    const int aRow1 = aRow0 + 16;

    v8f acc[2][4];
#pragma unroll
    for (int m = 0; m < 2; ++m)
#pragma unroll
        for (int n = 0; n < 4; ++n) acc[m][n] = ZERO8;

    for (int k0 = 0; k0 < D_MODEL; k0 += 32) {
        __builtin_prefetch(&ctx[(size_t)aRow0 * D_MODEL + k0 + 256], 0, 3);
        v16bf a0 = load_a_bf16(ctx, D_MODEL, aRow0, k0, half);
        v16bf a1 = load_a_bf16(ctx, D_MODEL, aRow1, k0, half);
#pragma unroll
        for (int n = 0; n < 4; ++n) {
            const int bCol = (tng * 4 + n) * 16 + lrow;
            v16bf b = load_b_nk(Wo, D_MODEL, bCol, k0, half);
            acc[0][n] = wmma_bf16(a0, b, acc[0][n]);
            acc[1][n] = wmma_bf16(a1, b, acc[1][n]);
        }
    }

    const float sfac = stab[0];
#pragma unroll
    for (int mt = 0; mt < 2; ++mt) {
#pragma unroll
        for (int n = 0; n < 4; ++n) {
            const int ncol = (tng * 4 + n) * 16 + lrow;
            const float bn = bo[ncol];
#pragma unroll
            for (int v = 0; v < 8; ++v) {
                const int m = tmg * 32 + mt * 16 + v + 8 * half;
                out[(size_t)m * D_MODEL + ncol] = (acc[mt][n][v] + bn) * sfac;
            }
        }
    }
}

// ---------------------------------------------------------------------------
// Launch
// ---------------------------------------------------------------------------
extern "C" void kernel_launch(void* const* d_in, const int* in_sizes, int n_in,
                              void* d_out, int out_size, void* d_ws, size_t ws_size,
                              hipStream_t stream) {
    const float* q     = (const float*)d_in[0];
    const float* k     = (const float*)d_in[1];
    const float* v     = (const float*)d_in[2];
    const float* phase = (const float*)d_in[3];
    const float* Wq    = (const float*)d_in[4];
    const float* bq    = (const float*)d_in[5];
    const float* Wk    = (const float*)d_in[6];
    const float* bk    = (const float*)d_in[7];
    const float* Wv    = (const float*)d_in[8];
    const float* bv    = (const float*)d_in[9];
    const float* Wo    = (const float*)d_in[10];
    const float* bo    = (const float*)d_in[11];
    const float* coupling   = (const float*)d_in[12];  // [1,H,1,1]
    const float* attn_scale = (const float*)d_in[13];
    const float* stab       = (const float*)d_in[14];

    float* out_main = (float*)d_out;                               // [B,S,D]
    float* attn_w   = out_main + (size_t)BATCH * SEQ * D_MODEL;    // [B,H,S,S]

    // Workspace carve-up
    char* ws = (char*)d_ws;
    const size_t W_ELEMS = (size_t)D_MODEL * D_MODEL;      // 1M
    const size_t A_ELEMS = (size_t)ROWS * D_MODEL;         // 4M
    const size_t ATT_ELEMS = (size_t)BATCH * NUM_HEADS * SEQ * SEQ; // 134M
    bf16_t* Wq_bf = (bf16_t*)ws;  ws += W_ELEMS * sizeof(bf16_t);
    bf16_t* Wk_bf = (bf16_t*)ws;  ws += W_ELEMS * sizeof(bf16_t);
    bf16_t* Wv_bf = (bf16_t*)ws;  ws += W_ELEMS * sizeof(bf16_t);
    bf16_t* Wo_bf = (bf16_t*)ws;  ws += W_ELEMS * sizeof(bf16_t);
    bf16_t* xq    = (bf16_t*)ws;  ws += A_ELEMS * sizeof(bf16_t);
    bf16_t* xk    = (bf16_t*)ws;  ws += A_ELEMS * sizeof(bf16_t);
    bf16_t* xv    = (bf16_t*)ws;  ws += A_ELEMS * sizeof(bf16_t);
    bf16_t* qh    = (bf16_t*)ws;  ws += A_ELEMS * sizeof(bf16_t);
    bf16_t* kh    = (bf16_t*)ws;  ws += A_ELEMS * sizeof(bf16_t);
    bf16_t* vhT   = (bf16_t*)ws;  ws += A_ELEMS * sizeof(bf16_t);
    bf16_t* ctx   = (bf16_t*)ws;  ws += A_ELEMS * sizeof(bf16_t);
    float*  sf    = (float*)ws;   ws += 256;               // 32 floats, padded
    const size_t base_bytes = (size_t)(ws - (char*)d_ws);
    // Optional bf16 attn copy (halves the context-GEMM read stream)
    const bool useBfAttn = ws_size >= base_bytes + ATT_ELEMS * sizeof(bf16_t);
    bf16_t* attn_bf = useBfAttn ? (bf16_t*)ws : nullptr;

    const dim3 blk(256);

    // 1) fp32 -> bf16 conversions + score-scale precompute
    cvt_f32_bf16_kernel<<<(int)(W_ELEMS / 256), blk, 0, stream>>>(Wq, Wq_bf, (int)W_ELEMS);
    cvt_f32_bf16_kernel<<<(int)(W_ELEMS / 256), blk, 0, stream>>>(Wk, Wk_bf, (int)W_ELEMS);
    cvt_f32_bf16_kernel<<<(int)(W_ELEMS / 256), blk, 0, stream>>>(Wv, Wv_bf, (int)W_ELEMS);
    cvt_f32_bf16_kernel<<<(int)(W_ELEMS / 256), blk, 0, stream>>>(Wo, Wo_bf, (int)W_ELEMS);
    cvt_f32_bf16_kernel<<<(int)(A_ELEMS / 256), blk, 0, stream>>>(q, xq, (int)A_ELEMS);
    cvt_f32_bf16_kernel<<<(int)(A_ELEMS / 256), blk, 0, stream>>>(k, xk, (int)A_ELEMS);
    cvt_f32_bf16_kernel<<<(int)(A_ELEMS / 256), blk, 0, stream>>>(v, xv, (int)A_ELEMS);
    score_scale_kernel<<<1, 32, 0, stream>>>(phase, coupling, attn_scale, sf);

    // 2) projections: 128 M-groups * 16 N-groups = 2048 waves -> 256 blocks
    proj_kernel<0><<<256, blk, 0, stream>>>(xq, Wq_bf, bq, qh);
    proj_kernel<0><<<256, blk, 0, stream>>>(xk, Wk_bf, bk, kh);
    proj_kernel<1><<<256, blk, 0, stream>>>(xv, Wv_bf, bv, vhT);

    // 3) scores: B*H * 64 q-groups * 32 k-groups = 65536 waves -> 8192 blocks
    scores_kernel<<<8192, blk, 0, stream>>>(qh, kh, sf, attn_w);

    // 4) softmax: one block per row (B*H*S = 65536 rows)
    softmax_kernel<<<BATCH * NUM_HEADS * SEQ, blk, 0, stream>>>(attn_w, attn_bf);

    // 5) context: B*H * 64 M-groups = 2048 waves -> 256 blocks
    if (useBfAttn)
        context_kernel<true><<<256, blk, 0, stream>>>(attn_w, attn_bf, vhT, ctx);
    else
        context_kernel<false><<<256, blk, 0, stream>>>(attn_w, attn_bf, vhT, ctx);

    // 6) output projection: 256 blocks
    out_proj_kernel<<<256, blk, 0, stream>>>(ctx, Wo_bf, bo, stab, out_main);
}
